// SGCN_34995393528531
// MI455X (gfx1250) — matrix operand order repaired and checked by reading, not compiled, not run on hardware
//
#include <hip/hip_runtime.h>
#include <hip/hip_bf16.h>

// ---------------------------------------------------------------------------
// Types for CDNA5 WMMA (wave32): D[16x16,f32] = A[16x32,bf16] * B[32x16,bf16] + C
// ---------------------------------------------------------------------------
typedef __bf16 bf16_t;
typedef __attribute__((ext_vector_type(16))) __bf16 v16bf;
typedef __attribute__((ext_vector_type(8)))  float  v8f;

__device__ __forceinline__ bf16_t f2bf(float f) {
    unsigned u = __builtin_bit_cast(unsigned, f);
    unsigned short s = (unsigned short)((u + 0x7FFFu + ((u >> 16) & 1u)) >> 16);
    return __builtin_bit_cast(bf16_t, s);
}

// Load one B-fragment (32x16 bf16 tile) pre-packed so each lane's 16 values are
// contiguous: lanes 0-15 hold N=lane, K=0..15; lanes 16-31 hold N=lane-16, K=16..31.
__device__ __forceinline__ v16bf load_bfrag(const bf16_t* __restrict__ tile, int lane) {
    return *(const v16bf*)(tile + lane * 16);
}

// Load NK A-fragments (16x32 bf16 each) from an LDS row-major [16 x stride] tile.
// A layout: lanes 0-15 -> M=lane, K in k0+0..15;
//           lanes 16-31 -> M=lane-16, K in k0+16..31.
template <int NK>
__device__ __forceinline__ void load_afrags(const bf16_t* __restrict__ lds, int stride,
                                            int lane, v16bf* af) {
    const bf16_t* row = lds + (lane & 15) * stride + ((lane >> 4) * 16);
#pragma unroll
    for (int kt = 0; kt < NK; ++kt) {
        const bf16_t* p = row + kt * 32;
        v16bf a;
#pragma unroll
        for (int v = 0; v < 8; ++v) a[v] = p[v];          // K = k0+hi*16 + 0..7
#pragma unroll
        for (int v = 0; v < 8; ++v) a[8 + v] = p[8 + v];  // K = k0+hi*16 + 8..15
        af[kt] = a;
    }
}

#define WMMA_BF16(A, B, C) \
    __builtin_amdgcn_wmma_f32_16x16x32_bf16(false, (A), false, (B), (short)0, (C), false, false)

#define LSTR 264   // 256-wide LDS rows, padded (528B row stride -> no bank clash)
#define TSTR 136   // 128-wide LDS rows, padded

// ---------------------------------------------------------------------------
// Weight pre-pack: f32 [K,128] -> bf16 B-fragment tiles [(K/32)*8][512]
// ---------------------------------------------------------------------------
__global__ void pack_b_kernel(const float* __restrict__ W, bf16_t* __restrict__ out, int K) {
    int idx = blockIdx.x * blockDim.x + threadIdx.x;
    if (idx >= K * 128) return;
    int kg = idx >> 7, ng = idx & 127;
    int kt = kg >> 5, kk = kg & 31;
    int nt = ng >> 4, nn = ng & 15;
    int lane = ((kk & 16) ? 16 : 0) + nn;
    int q = kk & 15;
    out[(size_t)(kt * 8 + nt) * 512 + lane * 16 + q] = f2bf(W[idx]);
}

__global__ void fill_kernel(float* __restrict__ p, long n, float v) {
    long i = (long)blockIdx.x * blockDim.x + threadIdx.x;
    if (i < n) p[i] = v;
}

__global__ void deg_kernel(const int* __restrict__ eidx, long E, float* __restrict__ deg) {
    long e = (long)blockIdx.x * blockDim.x + threadIdx.x;
    if (e < E) atomicAdd(&deg[eidx[E + e]], 1.0f);
}

__global__ void invdeg_kernel(float* __restrict__ d, int N) {
    int i = blockIdx.x * blockDim.x + threadIdx.x;
    if (i < N) { float v = d[i]; d[i] = v > 0.f ? 1.f / v : 0.f; }
}

// ---------------------------------------------------------------------------
// Node encoder: h = relu(x*w1+b1) @ w2 + b2   (1->128->128, f32 VALU; tiny)
// ---------------------------------------------------------------------------
__global__ __launch_bounds__(128) void node_mlp_kernel(
    const float* __restrict__ x, const float* __restrict__ w1, const float* __restrict__ b1,
    const float* __restrict__ w2, const float* __restrict__ b2,
    float* __restrict__ h, bf16_t* __restrict__ hbf, int N) {
    int i = blockIdx.x;
    if (i >= N) return;
    int t = threadIdx.x;
    __shared__ float a1[128];
    float a = x[i] * w1[t] + b1[t];
    a1[t] = a > 0.f ? a : 0.f;
    __syncthreads();
    float acc = b2[t];
#pragma unroll 8
    for (int k = 0; k < 128; ++k) acc += a1[k] * w2[k * 128 + t];
    h[(long)i * 128 + t] = acc;
    hbf[(long)i * 128 + t] = f2bf(acc);
}

// ---------------------------------------------------------------------------
// Fused per-edge pipeline (1 wave = 16 edges):
//   dist = pos[src]-pos[dst]
//   sp   = relu(dist@w31+b31) @ w32 + b32                       (WMMA x32)
//   msg  = relu([h[src], sp]@w21+b21) @ w22 + b22               (WMMA x96)
//   atomicAdd agg[dst] += msg
// ---------------------------------------------------------------------------
__global__ __launch_bounds__(128) void edge_kernel(
    const int* __restrict__ eidx, long E,
    const float* __restrict__ pos,
    const bf16_t* __restrict__ hbf,
    float* __restrict__ agg,
    const float* __restrict__ w31, const float* __restrict__ b31,
    const bf16_t* __restrict__ pw32, const float* __restrict__ b32,
    const bf16_t* __restrict__ pw21, const float* __restrict__ b21,
    const bf16_t* __restrict__ pw22, const float* __restrict__ b22) {
    __shared__ __align__(16) bf16_t s_in[4][16 * LSTR];  // [h[src] | spatial] bf16
    __shared__ __align__(16) bf16_t s_t[4][16 * TSTR];   // intermediate 128-wide
    __shared__ float s_dist[4][16][4];
    __shared__ int s_src[4][16], s_dst[4][16];

    const int wv = threadIdx.x >> 5, lane = threadIdx.x & 31;
    const long e0 = ((long)blockIdx.x * 4 + wv) * 16;
    if (e0 >= E) return;  // wave-uniform exit; no block barriers used below

    bf16_t* sin = s_in[wv];
    bf16_t* stt = s_t[wv];
    const int coln = lane & 15, rb = (lane >> 4) * 8;

    // --- edge indices + dist -------------------------------------------------
    if (lane < 16) {
        long e = e0 + lane;
        if (e >= E) e = E - 1;
        int s = eidx[e], d = eidx[E + e];
        s_src[wv][lane] = s;
        s_dst[wv][lane] = d;
        s_dist[wv][lane][0] = pos[(long)s * 3 + 0] - pos[(long)d * 3 + 0];
        s_dist[wv][lane][1] = pos[(long)s * 3 + 1] - pos[(long)d * 3 + 1];
        s_dist[wv][lane][2] = pos[(long)s * 3 + 2] - pos[(long)d * 3 + 2];
    }

    // --- gather h[src] (bf16) into cols 0..127 ------------------------------
#pragma unroll 4
    for (int e = 0; e < 16; ++e) {
        int s = s_src[wv][e];
        const uint2* gp = (const uint2*)(hbf + (long)s * 128);
        *(uint2*)(&sin[e * LSTR + lane * 4]) = gp[lane];
    }

    // --- spatial layer 1 (K=3, VALU): relu(dist@w31+b31) -> stt -------------
    for (int idx = lane; idx < 2048; idx += 32) {
        int e = idx >> 7, j = idx & 127;
        float a = b31[j] + s_dist[wv][e][0] * w31[j]
                         + s_dist[wv][e][1] * w31[128 + j]
                         + s_dist[wv][e][2] * w31[256 + j];
        stt[e * TSTR + j] = f2bf(a > 0.f ? a : 0.f);
    }

    // --- spatial layer 2: [16,128]@[128,128] -> cols 128..255 of sin --------
    {
        v16bf af[4];
        load_afrags<4>(stt, TSTR, lane, af);
#pragma unroll
        for (int nt = 0; nt < 8; ++nt) {
            v8f c = {0.f, 0.f, 0.f, 0.f, 0.f, 0.f, 0.f, 0.f};
#pragma unroll
            for (int kt = 0; kt < 4; ++kt)
                c = WMMA_BF16(af[kt], load_bfrag(pw32 + (kt * 8 + nt) * 512, lane), c);
            int col = nt * 16 + coln;
            float bb = b32[col];
#pragma unroll
            for (int r = 0; r < 8; ++r)
                sin[(rb + r) * LSTR + 128 + col] = f2bf(c[r] + bb);
        }
    }

    // --- message layer 1: [16,256]@[256,128], relu -> stt --------------------
    {
        v16bf af[8];
        load_afrags<8>(sin, LSTR, lane, af);
#pragma unroll
        for (int nt = 0; nt < 8; ++nt) {
            v8f c = {0.f, 0.f, 0.f, 0.f, 0.f, 0.f, 0.f, 0.f};
#pragma unroll
            for (int kt = 0; kt < 8; ++kt)
                c = WMMA_BF16(af[kt], load_bfrag(pw21 + (kt * 8 + nt) * 512, lane), c);
            int col = nt * 16 + coln;
            float bb = b21[col];
#pragma unroll
            for (int r = 0; r < 8; ++r) {
                float v = c[r] + bb;
                stt[(rb + r) * TSTR + col] = f2bf(v > 0.f ? v : 0.f);
            }
        }
    }

    // --- message layer 2 + mean-aggregation scatter -------------------------
    {
        v16bf af[4];
        load_afrags<4>(stt, TSTR, lane, af);
#pragma unroll
        for (int nt = 0; nt < 8; ++nt) {
            v8f c = {0.f, 0.f, 0.f, 0.f, 0.f, 0.f, 0.f, 0.f};
#pragma unroll
            for (int kt = 0; kt < 4; ++kt)
                c = WMMA_BF16(af[kt], load_bfrag(pw22 + (kt * 8 + nt) * 512, lane), c);
            int col = nt * 16 + coln;
            float bb = b22[col];
#pragma unroll
            for (int r = 0; r < 8; ++r) {
                int row = rb + r;
                if (e0 + row < E)
                    atomicAdd(&agg[(long)s_dst[wv][row] * 128 + col], c[r] + bb);
            }
        }
    }
}

// ---------------------------------------------------------------------------
// Per-node update (1 wave = 16 nodes):
//   h = relu( relu([agg*inv_deg, h]@w11+b11) @ w12 + b12 )
// ---------------------------------------------------------------------------
__global__ __launch_bounds__(128) void node_update_kernel(
    const float* __restrict__ agg, const float* __restrict__ invd,
    float* __restrict__ h, bf16_t* __restrict__ hbf,
    const bf16_t* __restrict__ pw11, const float* __restrict__ b11,
    const bf16_t* __restrict__ pw12, const float* __restrict__ b12, int N) {
    __shared__ __align__(16) bf16_t s_in[4][16 * LSTR];
    __shared__ __align__(16) bf16_t s_t[4][16 * TSTR];

    const int wv = threadIdx.x >> 5, lane = threadIdx.x & 31;
    const long n0 = ((long)blockIdx.x * 4 + wv) * 16;
    if (n0 >= N) return;
    bf16_t* sin = s_in[wv];
    bf16_t* stt = s_t[wv];
    const int coln = lane & 15, rb = (lane >> 4) * 8;

    for (int idx = lane; idx < 2048; idx += 32) {
        int e = idx >> 7, c = idx & 127;
        long nn = n0 + e;
        if (nn >= N) nn = N - 1;
        sin[e * LSTR + c] = f2bf(agg[nn * 128 + c] * invd[nn]);
        sin[e * LSTR + 128 + c] = hbf[nn * 128 + c];
    }

    {   // layer 1: [16,256]@[256,128], relu -> stt
        v16bf af[8];
        load_afrags<8>(sin, LSTR, lane, af);
#pragma unroll
        for (int nt = 0; nt < 8; ++nt) {
            v8f c = {0.f, 0.f, 0.f, 0.f, 0.f, 0.f, 0.f, 0.f};
#pragma unroll
            for (int kt = 0; kt < 8; ++kt)
                c = WMMA_BF16(af[kt], load_bfrag(pw11 + (kt * 8 + nt) * 512, lane), c);
            int col = nt * 16 + coln;
            float bb = b11[col];
#pragma unroll
            for (int r = 0; r < 8; ++r) {
                float v = c[r] + bb;
                stt[(rb + r) * TSTR + col] = f2bf(v > 0.f ? v : 0.f);
            }
        }
    }
    {   // layer 2: [16,128]@[128,128], outer relu -> h (f32) and hbf (bf16)
        v16bf af[4];
        load_afrags<4>(stt, TSTR, lane, af);
#pragma unroll
        for (int nt = 0; nt < 8; ++nt) {
            v8f c = {0.f, 0.f, 0.f, 0.f, 0.f, 0.f, 0.f, 0.f};
#pragma unroll
            for (int kt = 0; kt < 4; ++kt)
                c = WMMA_BF16(af[kt], load_bfrag(pw12 + (kt * 8 + nt) * 512, lane), c);
            int col = nt * 16 + coln;
            float bb = b12[col];
#pragma unroll
            for (int r = 0; r < 8; ++r) {
                long nn = n0 + rb + r;
                if (nn < N) {
                    float v = c[r] + bb;
                    v = v > 0.f ? v : 0.f;
                    h[nn * 128 + col] = v;
                    hbf[nn * 128 + col] = f2bf(v);
                }
            }
        }
    }
}

// ---------------------------------------------------------------------------
// Readout: y = relu(h@lin1+b1)@lin2+b2 ; out[batch[i]] += y   (tiny, VALU)
// ---------------------------------------------------------------------------
__global__ __launch_bounds__(64) void readout_kernel(
    const float* __restrict__ h, const int* __restrict__ batch,
    const float* __restrict__ l1w, const float* __restrict__ l1b,
    const float* __restrict__ l2w, const float* __restrict__ l2b,
    float* __restrict__ out, int N) {
    int i = blockIdx.x;
    if (i >= N) return;
    int t = threadIdx.x;  // 64
    __shared__ float red[64];
    float acc = l1b[t];
#pragma unroll 8
    for (int k = 0; k < 128; ++k) acc += h[(long)i * 128 + k] * l1w[k * 64 + t];
    acc = acc > 0.f ? acc : 0.f;
    red[t] = acc * l2w[t];
    __syncthreads();
    if (t == 0) {
        float s = l2b[0];
        for (int k = 0; k < 64; ++k) s += red[k];
        atomicAdd(&out[batch[i]], s);
    }
}

// ---------------------------------------------------------------------------
extern "C" void kernel_launch(void* const* d_in, const int* in_sizes, int n_in,
                              void* d_out, int out_size, void* d_ws, size_t ws_size,
                              hipStream_t stream) {
    const int H = 128, L = 3;
    const float* x        = (const float*)d_in[0];
    const float* pos      = (const float*)d_in[1];
    const int*   eidx     = (const int*)d_in[2];
    const int*   batch    = (const int*)d_in[3];
    const float* node_w1  = (const float*)d_in[4];
    const float* node_b1  = (const float*)d_in[5];
    const float* node_w2  = (const float*)d_in[6];
    const float* node_b2  = (const float*)d_in[7];
    const float* mlp1_w1  = (const float*)d_in[8];
    const float* mlp1_b1  = (const float*)d_in[9];
    const float* mlp1_w2  = (const float*)d_in[10];
    const float* mlp1_b2  = (const float*)d_in[11];
    const float* mlp2_w1  = (const float*)d_in[12];
    const float* mlp2_b1  = (const float*)d_in[13];
    const float* mlp2_w2  = (const float*)d_in[14];
    const float* mlp2_b2  = (const float*)d_in[15];
    const float* mlp3_w1  = (const float*)d_in[16];
    const float* mlp3_b1  = (const float*)d_in[17];
    const float* mlp3_w2  = (const float*)d_in[18];
    const float* mlp3_b2  = (const float*)d_in[19];
    const float* lin1_w   = (const float*)d_in[20];
    const float* lin1_b   = (const float*)d_in[21];
    const float* lin2_w   = (const float*)d_in[22];
    const float* lin2_b   = (const float*)d_in[23];

    const int  N = in_sizes[0];
    const long E = (long)in_sizes[2] / 2;

    // ---- workspace carve-up -------------------------------------------------
    char* p = (char*)d_ws;
    auto alloc = [&](size_t b) { void* r = (void*)p; p += (b + 255) & ~(size_t)255; return r; };
    float*  h    = (float*)alloc((size_t)N * H * sizeof(float));      // 25.6 MB
    float*  agg  = (float*)alloc((size_t)N * H * sizeof(float));      // 25.6 MB
    bf16_t* hbf  = (bf16_t*)alloc((size_t)N * H * sizeof(bf16_t));    // 12.8 MB
    float*  invd = (float*)alloc((size_t)N * sizeof(float));
    const size_t SZ_HH  = 16384;   // 128x128 bf16 elems
    const size_t SZ_2HH = 32768;   // 256x128 bf16 elems
    bf16_t* pk   = (bf16_t*)alloc((size_t)L * (SZ_HH * 3 + SZ_2HH * 2) * sizeof(bf16_t));
    bf16_t* p_w32 = pk;
    bf16_t* p_w21 = p_w32 + L * SZ_HH;
    bf16_t* p_w22 = p_w21 + L * SZ_2HH;
    bf16_t* p_w11 = p_w22 + L * SZ_HH;
    bf16_t* p_w12 = p_w11 + L * SZ_2HH;

    // ---- pack weights into B-fragment layout (bf16) ------------------------
    for (int l = 0; l < L; ++l) {
        pack_b_kernel<<<(H * H + 255) / 256, 256, 0, stream>>>(mlp3_w2 + (size_t)l * H * H, p_w32 + l * SZ_HH, H);
        pack_b_kernel<<<(2 * H * H + 255) / 256, 256, 0, stream>>>(mlp2_w1 + (size_t)l * 2 * H * H, p_w21 + l * SZ_2HH, 2 * H);
        pack_b_kernel<<<(H * H + 255) / 256, 256, 0, stream>>>(mlp2_w2 + (size_t)l * H * H, p_w22 + l * SZ_HH, H);
        pack_b_kernel<<<(2 * H * H + 255) / 256, 256, 0, stream>>>(mlp1_w1 + (size_t)l * 2 * H * H, p_w11 + l * SZ_2HH, 2 * H);
        pack_b_kernel<<<(H * H + 255) / 256, 256, 0, stream>>>(mlp1_w2 + (size_t)l * H * H, p_w12 + l * SZ_HH, H);
    }

    // ---- degrees ------------------------------------------------------------
    fill_kernel<<<(N + 255) / 256, 256, 0, stream>>>(invd, N, 0.f);
    deg_kernel<<<(int)((E + 255) / 256), 256, 0, stream>>>(eidx, E, invd);
    invdeg_kernel<<<(N + 255) / 256, 256, 0, stream>>>(invd, N);

    // ---- node encoder -------------------------------------------------------
    node_mlp_kernel<<<N, 128, 0, stream>>>(x, node_w1, node_b1, node_w2, node_b2, h, hbf, N);

    // ---- message-passing layers --------------------------------------------
    const long etiles = (E + 15) / 16;
    const int eblocks = (int)((etiles + 3) / 4);
    const int ntiles = (N + 15) / 16;
    const int nblocks = (ntiles + 3) / 4;
    for (int l = 0; l < L; ++l) {
        fill_kernel<<<(int)(((long)N * H + 255) / 256), 256, 0, stream>>>(agg, (long)N * H, 0.f);
        edge_kernel<<<eblocks, 128, 0, stream>>>(
            eidx, E, pos, hbf, agg,
            mlp3_w1 + (size_t)l * 3 * H, mlp3_b1 + (size_t)l * H,
            p_w32 + l * SZ_HH, mlp3_b2 + (size_t)l * H,
            p_w21 + l * SZ_2HH, mlp2_b1 + (size_t)l * H,
            p_w22 + l * SZ_HH, mlp2_b2 + (size_t)l * H);
        node_update_kernel<<<nblocks, 128, 0, stream>>>(
            agg, invd, h, hbf,
            p_w11 + l * SZ_2HH, mlp1_b1 + (size_t)l * H,
            p_w12 + l * SZ_HH, mlp1_b2 + (size_t)l * H, N);
    }

    // ---- readout ------------------------------------------------------------
    fill_kernel<<<(out_size + 255) / 256, 256, 0, stream>>>((float*)d_out, out_size, 0.f);
    readout_kernel<<<N, 64, 0, stream>>>(h, batch, lin1_w, lin1_b, lin2_w, lin2_b,
                                         (float*)d_out, N);
    (void)n_in; (void)ws_size;
}
